// SinkhornOT_58256936403165
// MI455X (gfx1250) — compile-verified, matrix operand order checked by reading, not run on hardware
//
#include <hip/hip_runtime.h>
#include <hip/hip_bf16.h>
#include <stdint.h>

// ---------------- problem constants ----------------
constexpr int Bb   = 128;
constexpr int Nn   = 512;
constexpr int SDd  = 256;
constexpr int FDd  = 256;
constexpr int PDd  = 128;
constexpr float EPSr = 0.1f;
constexpr int   NIT  = 30;

typedef __attribute__((ext_vector_type(16))) __bf16 bf16x16;
typedef __attribute__((ext_vector_type(8)))  __bf16 bf16x8;
typedef __attribute__((ext_vector_type(2)))  __bf16 bf16x2;
typedef __attribute__((ext_vector_type(8)))  float  v8f;

// K index mapping for 16-bit A/B fragments (ISA 7.12.2, 16-bit 16x32):
// lane<16,i<8 -> K=i ; lane<16,i>=8 -> K=i+8 ; lane>=16,i<8 -> K=i+8 ; lane>=16,i>=8 -> K=i+16
__device__ __forceinline__ int kmap(int i, int lanehi8) {
    return i + ((i & 8) ? 8 : 0) + lanehi8;
}

// ---------------------------------------------------------------------------
// Kernel 1: H = X @ W + b ; LayerNorm(P=128) ; L2-normalize ; emit bf16
// grid = (B*N)/16 blocks, 256 threads (8 waves); wave w owns output cols w*16..
// A-fragment fed by two 32B contiguous runs -> float4 (b128) loads.
// ---------------------------------------------------------------------------
__global__ __launch_bounds__(256) void proj_ln_kernel(
    const float* __restrict__ X, const float* __restrict__ W,
    const float* __restrict__ bias, const float* __restrict__ g,
    const float* __restrict__ beta, __bf16* __restrict__ out, int D)
{
    const int tid  = threadIdx.x;
    const int lane = tid & 31;
    const int wave = tid >> 5;
    const int r0   = blockIdx.x * 16;          // global row base
    const int l16  = lane & 15;
    const int hi8  = (lane & 16) ? 8 : 0;
    const int n0   = wave * 16;                // output col tile

    __shared__ float Hs[16][136];              // 16 x 128 (padded)

    v8f acc = {};
    const float* Xrow = X + (size_t)(r0 + l16) * D;
    for (int k0 = 0; k0 < D; k0 += 32) {
        // A: K = k0+hi8+{0..7} and k0+hi8+{16..23}  -> 4x float4
        const float4* Xv = reinterpret_cast<const float4*>(Xrow + k0 + hi8);
        float4 x0 = Xv[0], x1 = Xv[1], x2 = Xv[4], x3 = Xv[5];
        bf16x16 a;
        a[0]=(__bf16)x0.x; a[1]=(__bf16)x0.y; a[2]=(__bf16)x0.z; a[3]=(__bf16)x0.w;
        a[4]=(__bf16)x1.x; a[5]=(__bf16)x1.y; a[6]=(__bf16)x1.z; a[7]=(__bf16)x1.w;
        a[8]=(__bf16)x2.x; a[9]=(__bf16)x2.y; a[10]=(__bf16)x2.z; a[11]=(__bf16)x2.w;
        a[12]=(__bf16)x3.x; a[13]=(__bf16)x3.y; a[14]=(__bf16)x3.z; a[15]=(__bf16)x3.w;
        // B: per-lane column gather of W (coalesced across lanes)
        bf16x16 bm;
        #pragma unroll
        for (int i = 0; i < 16; ++i) {
            int k = k0 + kmap(i, hi8);
            bm[i] = (__bf16)W[(size_t)k * PDd + n0 + l16];
        }
        acc = __builtin_amdgcn_wmma_f32_16x16x32_bf16(false, a, false, bm,
                                                      (short)0, acc, false, false);
    }
    // D-fragment: vgpr r, lane -> row r + hi8, col l16
    #pragma unroll
    for (int r = 0; r < 8; ++r)
        Hs[r + hi8][n0 + l16] = acc[r] + bias[n0 + l16];
    __syncthreads();

    // LayerNorm + l2norm: thread t -> row t>>4, cols (t&15)+16j
    const int rrow = tid >> 4;
    const int cl   = tid & 15;
    float h[8], s1 = 0.f, s2 = 0.f;
    #pragma unroll
    for (int j = 0; j < 8; ++j) {
        float x = Hs[rrow][cl + 16 * j];
        h[j] = x; s1 += x; s2 += x * x;
    }
    #pragma unroll
    for (int off = 8; off >= 1; off >>= 1) {
        s1 += __shfl_xor(s1, off, 16);
        s2 += __shfl_xor(s2, off, 16);
    }
    float mu   = s1 * (1.f / PDd);
    float var  = s2 * (1.f / PDd) - mu * mu;       // biased variance (torch LN)
    float rstd = rsqrtf(var + 1e-5f);
    float y[8], q = 0.f;
    #pragma unroll
    for (int j = 0; j < 8; ++j) {
        int c = cl + 16 * j;
        float t = (h[j] - mu) * rstd * g[c] + beta[c];
        y[j] = t; q += t * t;
    }
    #pragma unroll
    for (int off = 8; off >= 1; off >>= 1) q += __shfl_xor(q, off, 16);
    float inv = 1.f / fmaxf(sqrtf(q), 1e-12f);
    __bf16* orow = out + (size_t)(r0 + rrow) * PDd;
    #pragma unroll
    for (int j = 0; j < 8; ++j) orow[cl + 16 * j] = (__bf16)(y[j] * inv);
}

// ---------------------------------------------------------------------------
// Kernel 2: K = exp(-clip(1 - Sp Fp^T, 0, 2)/eps)  per batch
// grid = B*32*4 blocks, 256 threads; wave owns one 16x16 tile (4 WMMAs, K=128)
// bf16 fragments loaded as 16B vectors.
// ---------------------------------------------------------------------------
__global__ __launch_bounds__(256) void cosk_kernel(
    const __bf16* __restrict__ Sp, const __bf16* __restrict__ Fp,
    float* __restrict__ Kout)
{
    const int tid  = threadIdx.x;
    const int lane = tid & 31;
    const int wave = tid >> 5;
    const int bi    = blockIdx.x;
    const int mb    = bi & 3;
    const int nb    = (bi >> 2) & 31;
    const int batch = bi >> 7;
    const int n0 = nb * 16;
    const int m0 = mb * 128 + wave * 16;
    const int l16 = lane & 15;
    const int hi8 = (lane & 16) ? 8 : 0;

    const __bf16* Sb = Sp + (size_t)batch * Nn * PDd + (size_t)(n0 + l16) * PDd + hi8;
    const __bf16* Fb = Fp + (size_t)batch * Nn * PDd + (size_t)(m0 + l16) * PDd + hi8;

    v8f acc = {};
    #pragma unroll
    for (int k0 = 0; k0 < PDd; k0 += 32) {
        bf16x8 alo = *reinterpret_cast<const bf16x8*>(Sb + k0);
        bf16x8 ahi = *reinterpret_cast<const bf16x8*>(Sb + k0 + 16);
        bf16x8 blo = *reinterpret_cast<const bf16x8*>(Fb + k0);
        bf16x8 bhi = *reinterpret_cast<const bf16x8*>(Fb + k0 + 16);
        bf16x16 a  = __builtin_shufflevector(alo, ahi, 0,1,2,3,4,5,6,7,8,9,10,11,12,13,14,15);
        bf16x16 bm = __builtin_shufflevector(blo, bhi, 0,1,2,3,4,5,6,7,8,9,10,11,12,13,14,15);
        acc = __builtin_amdgcn_wmma_f32_16x16x32_bf16(false, a, false, bm,
                                                      (short)0, acc, false, false);
    }
    float* Kb = Kout + (size_t)batch * Nn * Nn;
    #pragma unroll
    for (int r = 0; r < 8; ++r) {
        int nrow = n0 + r + hi8;
        float C = fminf(fmaxf(1.f - acc[r], 0.f), 2.f);
        Kb[(size_t)nrow * Nn + m0 + l16] = __expf(-C * (1.f / EPSr));
    }
}

// ---------------------------------------------------------------------------
// Kernel 3: 30 Sinkhorn iterations, one workgroup per batch.
// K (1 MB/batch, 134 MB total) stays resident in the 192 MB L2 across all 60
// matvec passes; u,v live in LDS. Matches reference: K^T u uses UNclipped u.
// ---------------------------------------------------------------------------
__global__ __launch_bounds__(512) void sinkhorn_kernel(
    const float* __restrict__ Kmat, float* __restrict__ u_out,
    float* __restrict__ v_out)
{
    __shared__ __align__(16) float su[Nn];
    __shared__ __align__(16) float sv[Nn];
    const int tid  = threadIdx.x;
    const int lane = tid & 31;
    const int wave = tid >> 5;        // 16 waves
    const int batch = blockIdx.x;
    const float* Kb = Kmat + (size_t)batch * Nn * Nn;
    const float rc = 1.f / Nn;

    sv[tid] = 1.f;
    __syncthreads();

    for (int it = 0; it < NIT; ++it) {
        // ---- u = r / (K v + 1e-8)   (wave-per-row, float4 per lane) ----
        for (int j = 0; j < 32; ++j) {
            int n = wave * 32 + j;
            const float* Kr = Kb + (size_t)n * Nn;
            float acc = 0.f;
            #pragma unroll
            for (int c = 0; c < 4; ++c) {
                float4 kv = *reinterpret_cast<const float4*>(Kr + lane * 4 + 128 * c);
                float4 vv = *reinterpret_cast<const float4*>(&sv[lane * 4 + 128 * c]);
                acc += kv.x * vv.x + kv.y * vv.y + kv.z * vv.z + kv.w * vv.w;
            }
            #pragma unroll
            for (int off = 16; off >= 1; off >>= 1) acc += __shfl_down(acc, off, 32);
            if (lane == 0) su[n] = rc / (acc + 1e-8f);   // unclipped for K^T u
        }
        __syncthreads();
        // ---- v = c / (K^T u + 1e-8)  (thread-per-column, coalesced) ----
        float acc = 0.f;
        for (int n = 0; n < Nn; ++n) {
            __builtin_prefetch(Kb + (size_t)(n + 8) * Nn + tid, 0, 1);
            acc += Kb[(size_t)n * Nn + tid] * su[n];
        }
        float vv = rc / (acc + 1e-8f);
        sv[tid] = fminf(fmaxf(vv, 1e-8f), 1e8f);         // clipped (carried)
        __syncthreads();
    }
    u_out[(size_t)batch * Nn + tid] = fminf(fmaxf(su[tid], 1e-8f), 1e8f);
    v_out[(size_t)batch * Nn + tid] = sv[tid];
}

// ---------------------------------------------------------------------------
// Kernel 4: P = u ∘ K ∘ v (written over K, in place) and F~ = P @ F via WMMA.
// F m-blocks (32x256 fp32, 32KB) are DMA'd to LDS with double-buffered
// global_load_async_to_lds_b128 (ASYNCcnt), overlapping DMA of block m0+32
// with the WMMAs of block m0. Row pad of 4 floats keeps b128 alignment and
// makes the 16-lane column gathers bank-conflict-free.
// grid = B*32 blocks (16-row strip), 256 threads; wave w owns cols w*16, w*16+128.
// ---------------------------------------------------------------------------
__global__ __launch_bounds__(256) void plan_ftilde_kernel(
    float* __restrict__ KP, const float* __restrict__ u,
    const float* __restrict__ v, const float* __restrict__ Fin,
    float* __restrict__ Fout)
{
    constexpr int FPAD = 260;                     // 256 + 4 (16B-aligned rows)
    __shared__ __align__(16) float  Fs[2][32][FPAD];
    __shared__ __align__(16) __bf16 Pt[16][40];   // 16 x 32 used (80B rows)

    const int tid  = threadIdx.x;
    const int lane = tid & 31;
    const int wave = tid >> 5;
    const int ntile = blockIdx.x & 31;
    const int batch = blockIdx.x >> 5;
    const int n0 = ntile * 16;
    const int l16 = lane & 15;
    const int hi8 = (lane & 16) ? 8 : 0;
    const int d0  = wave * 16;

    float* Kb = KP + (size_t)batch * Nn * Nn;
    const float* ub = u + (size_t)batch * Nn;
    const float* vb = v + (size_t)batch * Nn;
    const float* Fb = Fin + (size_t)batch * Nn * FDd;

    // async copy: 32 rows x 256 floats = 2048 16B chunks; 8 per thread
    const uint32_t lds0 = (uint32_t)(uintptr_t)(&Fs[0][0][0]);
    const uint32_t lds1 = (uint32_t)(uintptr_t)(&Fs[1][0][0]);
    const int chm = tid >> 3;            // wrong granularity? no: chunk decode below
    (void)chm;
    auto issue_async = [&](uint32_t ldsbase, int m0) {
        #pragma unroll
        for (int j = 0; j < 8; ++j) {
            int c  = tid + 256 * j;      // chunk id 0..2047
            int m  = c >> 6;             // row 0..31
            int dc = c & 63;             // 16B chunk within row
            uint64_t ga = (uint64_t)(uintptr_t)(Fb + (size_t)(m0 + m) * FDd + dc * 4);
            uint32_t la = ldsbase + (uint32_t)(m * FPAD * 4 + dc * 16);
            asm volatile("global_load_async_to_lds_b128 %0, %1, off"
                         :: "v"(la), "v"(ga) : "memory");
        }
    };

    // P tile: thread owns a float2 (rows 0..15 x cols 2t..2t+1 of 32)
    const int prow = tid >> 4;
    const int pml  = (tid & 15) * 2;
    const float up = ub[n0 + prow];

    issue_async(lds0, 0);

    v8f acc0 = {}, acc1 = {};
    for (int step = 0; step < Nn / 32; ++step) {
        const int m0  = step * 32;
        const int cur = step & 1;
        if (step + 1 < Nn / 32) issue_async(cur ? lds0 : lds1, m0 + 32);

        // form P tile, write P over K (same-thread read->write), stage bf16
        {
            size_t idx = (size_t)(n0 + prow) * Nn + m0 + pml;
            float2 kk = *reinterpret_cast<const float2*>(&Kb[idx]);
            float2 vv = *reinterpret_cast<const float2*>(&vb[m0 + pml]);
            float2 pp = make_float2(up * kk.x * vv.x, up * kk.y * vv.y);
            *reinterpret_cast<float2*>(&Kb[idx]) = pp;
            bf16x2 pb; pb[0] = (__bf16)pp.x; pb[1] = (__bf16)pp.y;
            *reinterpret_cast<bf16x2*>(&Pt[prow][pml]) = pb;
        }

        // current buffer's 8 async ops complete in-order; next step's 8 may remain
        if (step + 1 < Nn / 32) asm volatile("s_wait_asynccnt 0x8" ::: "memory");
        else                    asm volatile("s_wait_asynccnt 0x0" ::: "memory");
        __syncthreads();                 // Pt stores + all waves' DMA done

        const float (*Fsc)[FPAD] = Fs[cur];
        const __bf16* prowp = &Pt[l16][hi8];
        bf16x8 plo = *reinterpret_cast<const bf16x8*>(prowp);
        bf16x8 phi = *reinterpret_cast<const bf16x8*>(prowp + 16);
        bf16x16 a  = __builtin_shufflevector(plo, phi, 0,1,2,3,4,5,6,7,8,9,10,11,12,13,14,15);
        bf16x16 b0, b1;
        #pragma unroll
        for (int i = 0; i < 16; ++i) {
            int k = kmap(i, hi8);                        // local m within step
            b0[i] = (__bf16)Fsc[k][d0 + l16];
            b1[i] = (__bf16)Fsc[k][128 + d0 + l16];
        }
        acc0 = __builtin_amdgcn_wmma_f32_16x16x32_bf16(false, a, false, b0,
                                                       (short)0, acc0, false, false);
        acc1 = __builtin_amdgcn_wmma_f32_16x16x32_bf16(false, a, false, b1,
                                                       (short)0, acc1, false, false);
        __syncthreads();                 // protect Pt/Fs reuse next step
    }
    float* Fo = Fout + (size_t)batch * Nn * FDd;
    #pragma unroll
    for (int r = 0; r < 8; ++r) {
        int nrow = n0 + r + hi8;
        Fo[(size_t)nrow * FDd + d0 + l16]       = acc0[r];
        Fo[(size_t)nrow * FDd + 128 + d0 + l16] = acc1[r];
    }
}

// ---------------------------------------------------------------------------
extern "C" void kernel_launch(void* const* d_in, const int* in_sizes, int n_in,
                              void* d_out, int out_size, void* d_ws, size_t ws_size,
                              hipStream_t stream) {
    const float* S      = (const float*)d_in[0];
    const float* F      = (const float*)d_in[1];
    const float* W_s    = (const float*)d_in[2];
    const float* b_s    = (const float*)d_in[3];
    const float* g_s    = (const float*)d_in[4];
    const float* beta_s = (const float*)d_in[5];
    const float* W_f    = (const float*)d_in[6];
    const float* b_f    = (const float*)d_in[7];
    const float* g_f    = (const float*)d_in[8];
    const float* beta_f = (const float*)d_in[9];

    float* outP = (float*)d_out;                       // [B,N,N]  P (K staged here)
    float* outF = outP + (size_t)Bb * Nn * Nn;         // [B,N,FD] F_tilde

    // Stash bf16 Sp/Fp in the (not-yet-written) F_tilde region: 32 MB of 67 MB.
    __bf16* Sp = (__bf16*)outF;
    __bf16* Fp = Sp + (size_t)Bb * Nn * PDd;

    float* u = (float*)d_ws;                           // [B,N]
    float* v = u + (size_t)Bb * Nn;                    // [B,N]

    proj_ln_kernel<<<(Bb * Nn) / 16, 256, 0, stream>>>(S, W_s, b_s, g_s, beta_s, Sp, SDd);
    proj_ln_kernel<<<(Bb * Nn) / 16, 256, 0, stream>>>(F, W_f, b_f, g_f, beta_f, Fp, FDd);
    cosk_kernel<<<Bb * 32 * 4, 256, 0, stream>>>(Sp, Fp, outP);
    sinkhorn_kernel<<<Bb, 512, 0, stream>>>(outP, u, v);
    plan_ftilde_kernel<<<Bb * 32, 256, 0, stream>>>(outP, u, v, F, outF);
}
